// VectorQuantizer_78116865179754
// MI455X (gfx1250) — matrix operand order repaired and checked by reading, not compiled, not run on hardware
//
#include <hip/hip_runtime.h>
#include <hip/hip_bf16.h>

typedef __attribute__((ext_vector_type(8)))  __bf16 v8bf;
typedef __attribute__((ext_vector_type(16))) __bf16 v16bf;
typedef __attribute__((ext_vector_type(8)))  float  v8f;

#define NROWS 32768   // 8*4096 z vectors
#define NEMB  8192    // codebook entries
#define KDIM  256     // embedding dim
#define LDSSTRIDE 264 // padded bf16 stride (528B = 33*16B, kills bank conflicts)

// ---------------------------------------------------------------------------
// L2-normalize rows of [rows, 256] fp32 -> bf16. One wave32 per row.
// ---------------------------------------------------------------------------
__global__ void __launch_bounds__(256)
k_norm_bf16(const float* __restrict__ src, __bf16* __restrict__ dst, int rows) {
    const int wave = threadIdx.x >> 5;
    const int lane = threadIdx.x & 31;
    const int row  = blockIdx.x * 8 + wave;
    if (row >= rows) return;
    const float* p = src + (size_t)row * KDIM + lane * 8;
    float v[8];
    #pragma unroll
    for (int i = 0; i < 8; ++i) v[i] = p[i];
    float s = 0.f;
    #pragma unroll
    for (int i = 0; i < 8; ++i) s += v[i] * v[i];
    #pragma unroll
    for (int off = 16; off > 0; off >>= 1) s += __shfl_xor(s, off, 32);
    const float scale = 1.0f / fmaxf(sqrtf(s), 1e-12f);
    v8bf o;
    #pragma unroll
    for (int i = 0; i < 8; ++i) o[i] = (__bf16)(v[i] * scale);
    *(v8bf*)(dst + (size_t)row * KDIM + lane * 8) = o;
}

// ---------------------------------------------------------------------------
// scores = zf (NROWS x 256) . en^T (256 x NEMB), row-wise argmax.
// Block: 256 threads (8 waves), 128 rows. Wave w owns rows [w*16, w*16+16).
// Per 16x16 tile: preload 8 B fragments, then two independent 4-deep
// v_wmma_f32_16x16x32_bf16 accumulation chains (even/odd K), merged at end.
// ---------------------------------------------------------------------------
__global__ void __launch_bounds__(256)
k_argmax(const __bf16* __restrict__ zf, const __bf16* __restrict__ en,
         int* __restrict__ idxout) {
    extern __shared__ __bf16 smem[];
    __bf16* As = smem;                         // 128 x LDSSTRIDE
    __bf16* Bs = smem + 128 * LDSSTRIDE;       // 64  x LDSSTRIDE

    const int tid     = threadIdx.x;
    const int lane    = tid & 31;
    const int wave    = tid >> 5;
    const int rowBase = blockIdx.x * 128;

    // stage A tile: 128 rows x 256 bf16 (one shot)
    for (int i = tid; i < 128 * 32; i += 256) {
        const int r = i >> 5, c = (i & 31) * 8;
        *(v8bf*)(As + r * LDSSTRIDE + c) =
            *(const v8bf*)(zf + (size_t)(rowBase + r) * KDIM + c);
    }
    __syncthreads();

    // A fragments (ISA 16-bit A 16x32 layout): lane<16 -> K {0..7,16..23},
    // lane>=16 -> K {8..15,24..31}; row = lane&15.
    const int ar = lane & 15;
    const int ak = (lane >> 4) * 8;
    const __bf16* abase = As + (wave * 16 + ar) * LDSSTRIDE;
    v16bf afrag[8];
    #pragma unroll
    for (int kt = 0; kt < 8; ++kt) {
        union { v16bf v; v8bf h[2]; } u;
        u.h[0] = *(const v8bf*)(abase + kt * 32 + ak);
        u.h[1] = *(const v8bf*)(abase + kt * 32 + ak + 16);
        afrag[kt] = u.v;
    }

    float best[8];
    int   bidx[8];
    #pragma unroll
    for (int g = 0; g < 8; ++g) { best[g] = -3.0e38f; bidx[g] = 0; }

    // B fragment addressing (32x16 bf16 B): col = lane&15, K-base = (lane>>4)*8
    const int bc = lane & 15;
    const int bk = (lane >> 4) * 8;

    for (int chunk = 0; chunk < NEMB / 64; ++chunk) {
        __syncthreads();
        // stage 64 codebook rows x 256 bf16
        for (int i = tid; i < 64 * 32; i += 256) {
            const int r = i >> 5, c = (i & 31) * 8;
            *(v8bf*)(Bs + r * LDSSTRIDE + c) =
                *(const v8bf*)(en + (size_t)(chunk * 64 + r) * KDIM + c);
        }
        __syncthreads();

        // pull next chunk (32KB) toward the caches while this one computes
        if (chunk + 1 < NEMB / 64) {
            __builtin_prefetch(en + (size_t)(chunk + 1) * 64 * KDIM + tid * 64,
                               0, 1);
        }

        #pragma unroll
        for (int ct = 0; ct < 4; ++ct) {
            const __bf16* bbase = Bs + (ct * 16 + bc) * LDSSTRIDE + bk;
            // preload all 8 B fragments (16x ds_load_b128 as one burst)
            v16bf bfrag[8];
            #pragma unroll
            for (int kt = 0; kt < 8; ++kt) {
                union { v16bf v; v8bf h[2]; } u;
                u.h[0] = *(const v8bf*)(bbase + kt * 32);
                u.h[1] = *(const v8bf*)(bbase + kt * 32 + 16);
                bfrag[kt] = u.v;
            }
            // two independent accumulation chains for XDL pipelining
            v8f acc0 = {}, acc1 = {};
            #pragma unroll
            for (int kt = 0; kt < 8; kt += 2) {
                acc0 = __builtin_amdgcn_wmma_f32_16x16x32_bf16(
                    false, afrag[kt], false, bfrag[kt],
                    (short)0, acc0, false, false);
                acc1 = __builtin_amdgcn_wmma_f32_16x16x32_bf16(
                    false, afrag[kt + 1], false, bfrag[kt + 1],
                    (short)0, acc1, false, false);
            }
            const int col = chunk * 64 + ct * 16 + bc;
            #pragma unroll
            for (int g = 0; g < 8; ++g) {
                const float sc = acc0[g] + acc1[g];
                if (sc > best[g]) { best[g] = sc; bidx[g] = col; }
            }
        }
    }

    // reduce over the 16 lanes holding the same rows (xor 1,2,4,8 stays in half)
    #pragma unroll
    for (int s = 1; s < 16; s <<= 1) {
        #pragma unroll
        for (int g = 0; g < 8; ++g) {
            const float om = __shfl_xor(best[g], s, 32);
            const int   oi = __shfl_xor(bidx[g], s, 32);
            if (om > best[g] || (om == best[g] && oi < bidx[g])) {
                best[g] = om; bidx[g] = oi;
            }
        }
    }
    if ((lane & 15) == 0) {
        const int half = lane >> 4;
        #pragma unroll
        for (int g = 0; g < 8; ++g)
            idxout[rowBase + wave * 16 + half * 8 + g] = bidx[g];
    }
}

// ---------------------------------------------------------------------------
// Gather emb[idx], fp32 normalize both sides, emit z_qnorm + per-row sq-diff.
// One wave32 per row.
// ---------------------------------------------------------------------------
__global__ void __launch_bounds__(256)
k_gather(const float* __restrict__ z, const float* __restrict__ emb,
         const int* __restrict__ idx, float* __restrict__ out_zq,
         float* __restrict__ out_idx, float* __restrict__ rowsum) {
    const int wave = threadIdx.x >> 5;
    const int lane = threadIdx.x & 31;
    const int n    = blockIdx.x * 8 + wave;
    if (n >= NROWS) return;
    const int j = idx[n];
    const float* zp = z   + (size_t)n * KDIM + lane * 8;
    const float* ep = emb + (size_t)j * KDIM + lane * 8;
    float zv[8], ev[8];
    float sz = 0.f, se = 0.f;
    #pragma unroll
    for (int i = 0; i < 8; ++i) {
        zv[i] = zp[i]; ev[i] = ep[i];
        sz += zv[i] * zv[i]; se += ev[i] * ev[i];
    }
    #pragma unroll
    for (int off = 16; off > 0; off >>= 1) {
        sz += __shfl_xor(sz, off, 32);
        se += __shfl_xor(se, off, 32);
    }
    const float zs = 1.0f / fmaxf(sqrtf(sz), 1e-12f);
    const float es = 1.0f / fmaxf(sqrtf(se), 1e-12f);
    float d2 = 0.f;
    float* op = out_zq + (size_t)n * KDIM + lane * 8;
    #pragma unroll
    for (int i = 0; i < 8; ++i) {
        const float q = ev[i] * es;
        const float d = q - zv[i] * zs;
        d2 += d * d;
        op[i] = q;   // straight-through output == z_qnorm numerically
    }
    #pragma unroll
    for (int off = 16; off > 0; off >>= 1) d2 += __shfl_xor(d2, off, 32);
    if (lane == 0) {
        rowsum[n]  = d2;
        out_idx[n] = (float)j;
    }
}

// ---------------------------------------------------------------------------
// Deterministic reduction of 32768 row sums -> loss = 1.25 * mean(sq diff)
// ---------------------------------------------------------------------------
__global__ void __launch_bounds__(256)
k_loss(const float* __restrict__ rowsum, float* __restrict__ out_loss) {
    __shared__ float sh[256];
    float s = 0.f;
    for (int i = 0; i < NROWS / 256; ++i) s += rowsum[threadIdx.x + i * 256];
    sh[threadIdx.x] = s;
    __syncthreads();
    for (int w = 128; w > 0; w >>= 1) {
        if (threadIdx.x < w) sh[threadIdx.x] += sh[threadIdx.x + w];
        __syncthreads();
    }
    if (threadIdx.x == 0)
        out_loss[0] = sh[0] * 1.25f / ((float)NROWS * (float)KDIM);
}

extern "C" void kernel_launch(void* const* d_in, const int* in_sizes, int n_in,
                              void* d_out, int out_size, void* d_ws, size_t ws_size,
                              hipStream_t stream) {
    const float* z   = (const float*)d_in[0];   // [8,4096,256] fp32
    const float* emb = (const float*)d_in[1];   // [8192,256]  fp32
    float* out = (float*)d_out;
    // out layout: z_qnorm [0, 8388608) | loss @8388608 | idx [8388609, 8421377)
    char* ws = (char*)d_ws;
    __bf16* zf  = (__bf16*)ws;                                          // 16 MB
    __bf16* en  = (__bf16*)(ws + (size_t)NROWS * KDIM * 2);             //  4 MB
    int*    idx = (int*)  (ws + (size_t)NROWS * KDIM * 2
                              + (size_t)NEMB  * KDIM * 2);              // 128 KB
    float*  rsum = (float*)(idx + NROWS);                               // 128 KB

    k_norm_bf16<<<NROWS / 8, 256, 0, stream>>>(z,   zf, NROWS);
    k_norm_bf16<<<NEMB  / 8, 256, 0, stream>>>(emb, en, NEMB);

    const size_t lds = (size_t)(128 + 64) * LDSSTRIDE * sizeof(__bf16); // ~99 KB
    k_argmax<<<NROWS / 128, 256, lds, stream>>>(zf, en, idx);

    k_gather<<<NROWS / 8, 256, 0, stream>>>(z, emb, idx, out,
                                            out + 8388609, rsum);
    k_loss<<<1, 256, 0, stream>>>(rsum, out + 8388608);
}
